// ConcreteAutoencoder_3118146257105
// MI455X (gfx1250) — matrix operand order repaired and checked by reading, not compiled, not run on hardware
//
#include <hip/hip_runtime.h>
#include <math.h>

typedef __attribute__((ext_vector_type(16))) _Float16 v16h;
typedef __attribute__((ext_vector_type(8)))  _Float16 v8h;
typedef __attribute__((ext_vector_type(8)))  float    v8f;

#define BDIM 4096   // batch
#define NDIM 4096   // full state dim
#define RDIM 64     // r
#define NHAT 256
#define ENCI 768
#define ENCO 128
#define NLAM 16

// ---- workspace layout (bytes) ----
#define XT_OFF    0x000000u   // X_tilde  f32 4096x64   (1 MB)
#define YT_OFF    0x100000u   // Y_tilde  f32 4096x64   (1 MB)
#define G_OFF     0x200000u   // G0 = A^T A f32 128x128 (64 KB)
#define ATL_OFF   0x210000u   // A^T lhs  f32 128x64    (32 KB)
#define ATR_OFF   0x218000u   // A^T R0   f32 128x64    (32 KB)
#define ERRS_OFF  0x220000u   // 16 f32
#define SCAL_OFF  0x220100u   // [0]=||Yt||^2 [1]=||R0||^2 [2]=temp
#define WBUF_OFF  0x220200u   // 16 x (128x64) f32      (512 KB)
#define BPU_OFF   0x2A0400u   // U_l packed f16 B-frags (512 KB)
#define BPPHI_OFF 0x320400u   // phi packed f16         (32 KB)
#define BPAT_OFF  0x328400u   // A_tilde^T packed f16   (8 KB)
#define BPS_OFF   0x32A400u   // samples^T packed f16   (192 KB)
#define TNL_OFF   0x35A400u   // X_hat_nl^T f16 128x4096 (1 MB)
#define LHST_OFF  0x45A400u   // lhs^T f16 64x4096      (512 KB)
#define R0T_OFF   0x4DA400u   // R0^T  f16 64x4096      (512 KB)

__device__ __forceinline__ v8f wmma_f16(v16h a, v16h b, v8f c) {
  return __builtin_amdgcn_wmma_f32_16x16x32_f16(false, a, false, b, (short)0, c, false, false);
}

// A-fragment 16x32 f16 from row-major f32.  p = &row[kbase + 16*half].
// Lane layout: lane m & lane m+16 hold row m; per-half K runs
// [16h..16h+15]  ->  a[i]=K(16h+i).
__device__ __forceinline__ v16h afrag_f32(const float* p) {
  v16h a;
#pragma unroll
  for (int i = 0; i < 8; ++i) { a[i] = (_Float16)p[i]; a[i + 8] = (_Float16)p[8 + i]; }
  return a;
}
__device__ __forceinline__ v16h afrag_f16(const _Float16* p) {
  v16h a;
#pragma unroll
  for (int i = 0; i < 8; ++i) { a[i] = p[i]; a[i + 8] = p[8 + i]; }
  return a;
}

__device__ __forceinline__ void atomicAddF32(float* p, float v) {
  __hip_atomic_fetch_add(p, v, __ATOMIC_RELAXED, __HIP_MEMORY_SCOPE_AGENT);
}

// ---------------- zero accumulators ----------------
__global__ void k_zero(float* g, float* atl, float* atr, float* scal) {
  int i = blockIdx.x * 256 + threadIdx.x;
  if (i < ENCO * ENCO) g[i] = 0.f;
  if (i < ENCO * RDIM) { atl[i] = 0.f; atr[i] = 0.f; }
  if (i < 3) scal[i] = 0.f;
}

// ---------------- pack B-fragment buffers ----------------
// pack[kb][nb][lane][i] = B[32*kb + 16*(lane>>4) + i][16*nb + (lane&15)]
__global__ void k_pack(const float* U, const float* phi, const float* At, const float* lhs,
                       _Float16* bpu, _Float16* bpphi, _Float16* bpat, _Float16* lhsT) {
  int e = blockIdx.x * 256 + threadIdx.x;
  int task = blockIdx.y;
  if (task == 0) {                       // U_l: K=4096, N=64 -> kb=128, nb=4
    if (e >= 128 * 4 * 32 * 16) return;
    int i = e & 15, lane = (e >> 4) & 31, t = e >> 9;
    int nb = t & 3, kb = t >> 2;
    int k = 32 * kb + 16 * (lane >> 4) + i;
    int n = 16 * nb + (lane & 15);
    bpu[e] = (_Float16)U[(size_t)k * RDIM + n];
  } else if (task == 1) {                // phi: K=64, N=256 -> kb=2, nb=16
    if (e >= 2 * 16 * 32 * 16) return;
    int i = e & 15, lane = (e >> 4) & 31, t = e >> 9;
    int nb = t & 15, kb = t >> 4;
    int k = 32 * kb + 16 * (lane >> 4) + i;
    int n = 16 * nb + (lane & 15);
    bpphi[e] = (_Float16)phi[(size_t)k * NHAT + n];
  } else if (task == 2) {                // A_tilde^T: K=64, N=64 -> kb=2, nb=4
    if (e >= 2 * 4 * 32 * 16) return;
    int i = e & 15, lane = (e >> 4) & 31, t = e >> 9;
    int nb = t & 3, kb = t >> 2;
    int k = 32 * kb + 16 * (lane >> 4) + i;
    int n = 16 * nb + (lane & 15);
    bpat[e] = (_Float16)At[(size_t)n * RDIM + k];   // (A_tilde^T)[k][n] = A_tilde[n][k]
  } else {                               // lhs^T f16: lhsT[c][k] = lhs[k][c]
    if (e >= RDIM * BDIM) return;
    int c = e >> 12, k = e & 4095;
    lhsT[e] = (_Float16)lhs[(size_t)k * RDIM + c];
  }
}

// ---------------- projection GEMM: (M @ U_l + 1)/2 ----------------
__global__ void __launch_bounds__(256) k_proj(const float* X, const float* Y,
                                              const _Float16* bpu, float* Xt, float* Yt,
                                              float* scal) {
  const int tid = threadIdx.x, wave = tid >> 5, lane = tid & 31;
  const int half = lane >> 4, lcol = lane & 15;
  const int isY = blockIdx.y;
  const float* src = isY ? Y : X;
  float* dst = isY ? Yt : Xt;
  const int rowA = blockIdx.x * 128 + wave * 16 + lcol;
  v8f acc[4] = {};
  for (int kk = 0; kk < 128; ++kk) {
    v16h a = afrag_f32(src + (size_t)rowA * NDIM + kk * 32 + half * 16);
    const _Float16* bp = bpu + (size_t)kk * (4 * 32 * 16) + lane * 16;
#pragma unroll
    for (int nb = 0; nb < 4; ++nb) {
      v16h b = *(const v16h*)(bp + nb * 512);
      acc[nb] = wmma_f16(a, b, acc[nb]);
    }
  }
  const int rowBase = blockIdx.x * 128 + wave * 16 + half * 8;
  float ss = 0.f;
#pragma unroll
  for (int nb = 0; nb < 4; ++nb) {
#pragma unroll
    for (int v = 0; v < 8; ++v) {
      float val = acc[nb][v] * 0.5f + 0.5f;           // (x - MIN_TILDE)/scale
      dst[(size_t)(rowBase + v) * RDIM + nb * 16 + lcol] = val;
      ss += val * val;
    }
  }
  if (isY) atomicAddF32(&scal[0], ss);                // ||Y_tilde||^2
}

// ---------------- gumbel-softmax + argmax; pack samples^T ----------------
__global__ void __launch_bounds__(256) k_gumbel(const float* logits, const float* uni,
                                                const float* temperature, _Float16* bps,
                                                float* out, float* scal) {
  __shared__ float rv[256];
  __shared__ int ri[256];
  const int o = blockIdx.x, tid = threadIdx.x;
  float temp = fmaxf(0.1f, temperature[0] * 0.99f);
  float x[3], e[3];
  float lmax = -1e30f; int lidx = 0; float xmax = -1e30f;
#pragma unroll
  for (int j = 0; j < 3; ++j) {
    int k = tid + j * 256;
    float l = logits[(size_t)o * ENCI + k];
    float u = uni[(size_t)o * ENCI + k];
    float g = -logf(-logf(u));
    x[j] = (l + g) / temp;
    xmax = fmaxf(xmax, x[j]);
    if (l > lmax) { lmax = l; lidx = k; }
  }
  rv[tid] = xmax; __syncthreads();
  for (int s = 128; s > 0; s >>= 1) { if (tid < s) rv[tid] = fmaxf(rv[tid], rv[tid + s]); __syncthreads(); }
  float mx = rv[0]; __syncthreads();
  float esum = 0.f;
#pragma unroll
  for (int j = 0; j < 3; ++j) { e[j] = expf(x[j] - mx); esum += e[j]; }
  rv[tid] = esum; __syncthreads();
  for (int s = 128; s > 0; s >>= 1) { if (tid < s) rv[tid] += rv[tid + s]; __syncthreads(); }
  float inv = 1.f / rv[0]; __syncthreads();
  rv[tid] = lmax; ri[tid] = lidx; __syncthreads();
  for (int s = 128; s > 0; s >>= 1) {
    if (tid < s) {
      float ov = rv[tid + s]; int oi = ri[tid + s];
      if (ov > rv[tid] || (ov == rv[tid] && oi < ri[tid])) { rv[tid] = ov; ri[tid] = oi; }
    }
    __syncthreads();
  }
  if (tid == 0) {
    out[o] = (float)ri[0];                            // selected_idx
    if (o == 0) { scal[2] = temp; out[8321] = temp; } // temp output
  }
  // samples^T B-fragment pack: B[k][n] = samples[n][k], n = o
  const int nb = o >> 4, oc = o & 15;
#pragma unroll
  for (int j = 0; j < 3; ++j) {
    int k = tid + j * 256;
    int kb = k >> 5, h = (k >> 4) & 1, i = k & 15;
    int lane = 16 * h + oc;
    bps[((size_t)(kb * 8 + nb) * 32 + lane) * 16 + i] = (_Float16)(e[j] * inv);
  }
}

// ---------------- R0 = Xt @ A_tilde^T - Yt ; store R0^T f16, ||R0||^2 ----------------
__global__ void __launch_bounds__(256) k_r0(const float* Xt, const float* Yt,
                                            const _Float16* bpat, _Float16* r0t, float* scal) {
  const int tid = threadIdx.x, wave = tid >> 5, lane = tid & 31;
  const int half = lane >> 4, lcol = lane & 15;
  const int rowA = blockIdx.x * 128 + wave * 16 + lcol;
  v8f acc[4] = {};
#pragma unroll
  for (int kk = 0; kk < 2; ++kk) {
    v16h a = afrag_f32(Xt + (size_t)rowA * RDIM + kk * 32 + half * 16);
    const _Float16* bp = bpat + (size_t)kk * (4 * 32 * 16) + lane * 16;
#pragma unroll
    for (int nb = 0; nb < 4; ++nb) {
      v16h b = *(const v16h*)(bp + nb * 512);
      acc[nb] = wmma_f16(a, b, acc[nb]);
    }
  }
  const int rowBase = blockIdx.x * 128 + wave * 16 + half * 8;
  float ss = 0.f;
#pragma unroll
  for (int nb = 0; nb < 4; ++nb) {
    int col = nb * 16 + lcol;
    v8h pk;
#pragma unroll
    for (int v = 0; v < 8; ++v) {
      float r = acc[nb][v] - Yt[(size_t)(rowBase + v) * RDIM + col];
      ss += r * r;
      pk[v] = (_Float16)r;
    }
    *(v8h*)(r0t + (size_t)col * BDIM + rowBase) = pk;
  }
  atomicAddF32(&scal[1], ss);
}

// ---------------- fused X_hat -> library -> X_hat_nl^T ----------------
#define XH_STR 264
__global__ void __launch_bounds__(128) k_xhatnl(const float* Xt, const _Float16* bpphi,
                                                const _Float16* bps, _Float16* tnl) {
  __shared__ _Float16 xh[64 * XH_STR];
  const int tid = threadIdx.x, wave = tid >> 5, lane = tid & 31;
  const int half = lane >> 4, lcol = lane & 15;
  const int growA = blockIdx.x * 64 + wave * 16 + lcol;
  // phase 1: X_hat tile (64 x 256) = Xt @ phi, into LDS (f16)
  v16h aph[2];
#pragma unroll
  for (int kk = 0; kk < 2; ++kk)
    aph[kk] = afrag_f32(Xt + (size_t)growA * RDIM + kk * 32 + half * 16);
  for (int nb = 0; nb < 16; ++nb) {
    v8f acc = {};
#pragma unroll
    for (int kk = 0; kk < 2; ++kk) {
      v16h b = *(const v16h*)(bpphi + ((size_t)(kk * 16 + nb) * 32 + lane) * 16);
      acc = wmma_f16(aph[kk], b, acc);
    }
    const int lrowBase = wave * 16 + half * 8;
    const int col = nb * 16 + lcol;
#pragma unroll
    for (int v = 0; v < 8; ++v) xh[(lrowBase + v) * XH_STR + col] = (_Float16)acc[v];
  }
  __syncthreads();
  // phase 2: lib(X_hat) @ samples^T, K=768 (func uniform per 32-K step)
  v8f acc2[8] = {};
  const int lrow = wave * 16 + lcol;
  for (int kk = 0; kk < 24; ++kk) {
    const int func = kk >> 3;                 // 0:id 1:square 2:sin
    const int jb = (kk & 7) * 32;
    const _Float16* p = &xh[lrow * XH_STR + jb + half * 16];
    v16h a;
#pragma unroll
    for (int i = 0; i < 8; ++i) {
      float x0 = (float)p[i], x1 = (float)p[8 + i];
      if (func == 1) { x0 = x0 * x0; x1 = x1 * x1; }
      else if (func == 2) { x0 = __sinf(x0); x1 = __sinf(x1); }
      a[i] = (_Float16)x0; a[i + 8] = (_Float16)x1;
    }
    const _Float16* bp = bps + ((size_t)kk * 8 * 32 + lane) * 16;
#pragma unroll
    for (int nb = 0; nb < 8; ++nb) {
      v16h b = *(const v16h*)(bp + nb * 512);
      acc2[nb] = wmma_f16(a, b, acc2[nb]);
    }
  }
  const int growBase = blockIdx.x * 64 + wave * 16 + half * 8;
#pragma unroll
  for (int nb = 0; nb < 8; ++nb) {
    int col = nb * 16 + lcol;
    v8h pk;
#pragma unroll
    for (int v = 0; v < 8; ++v) pk[v] = (_Float16)acc2[nb][v];
    *(v8h*)(tnl + (size_t)col * BDIM + growBase) = pk;
  }
}

// ---------------- G0 = A^T A, AtL = A^T lhs, AtR = A^T R0 (split-K atomics) ----------------
__global__ void __launch_bounds__(256) k_atprod(const _Float16* tnl, const _Float16* lhsT,
                                                const _Float16* r0t, float* G, float* AtL,
                                                float* AtR) {
  const int tid = threadIdx.x, wave = tid >> 5, lane = tid & 31;
  const int half = lane >> 4, lcol = lane & 15;
  const int g = blockIdx.y;
  const int mrow = wave * 16 + lcol;
  const _Float16* arow = tnl + (size_t)mrow * BDIM;
  const _Float16* brow[8];
#pragma unroll
  for (int nb = 0; nb < 8; ++nb) {
    int cti = g * 8 + nb;
    const _Float16* src; int col;
    if (cti < 8)       { src = tnl;  col = cti * 16 + lcol; }
    else if (cti < 12) { src = lhsT; col = (cti - 8) * 16 + lcol; }
    else               { src = r0t;  col = (cti - 12) * 16 + lcol; }
    brow[nb] = src + (size_t)col * BDIM;
  }
  v8f acc[8] = {};
  const int k0 = blockIdx.x * 512;
  for (int kk = 0; kk < 16; ++kk) {
    int kb = k0 + kk * 32;
    v16h a = afrag_f16(arow + kb + half * 16);
#pragma unroll
    for (int nb = 0; nb < 8; ++nb) {
      v16h b = *(const v16h*)(brow[nb] + kb + half * 16);
      acc[nb] = wmma_f16(a, b, acc[nb]);
    }
  }
  const int mBase = wave * 16 + half * 8;
#pragma unroll
  for (int nb = 0; nb < 8; ++nb) {
    int cti = g * 8 + nb;
#pragma unroll
    for (int v = 0; v < 8; ++v) {
      int m = mBase + v;
      float val = acc[nb][v];
      if (cti < 8)       atomicAddF32(&G[m * ENCO + cti * 16 + lcol], val);
      else if (cti < 12) atomicAddF32(&AtL[m * RDIM + (cti - 8) * 16 + lcol], val);
      else               atomicAddF32(&AtR[m * RDIM + (cti - 12) * 16 + lcol], val);
    }
  }
}

// ---------------- per-lambda Gauss-Jordan solve + trace-trick error ----------------
__global__ void __launch_bounds__(192) k_solve(const float* G, const float* AtL, const float* AtR,
                                               const float* lam_vec, float* Wbuf,
                                               const float* scal, float* errs) {
  __shared__ float Gs[128 * 128];   // 64 KB
  const int tid = threadIdx.x;
  const float lam = lam_vec[blockIdx.x];
  float* W = Wbuf + (size_t)blockIdx.x * (ENCO * RDIM);
  for (int e = tid; e < ENCO * ENCO; e += 192) {
    float v = G[e];
    if ((e >> 7) == (e & 127)) v += lam;
    Gs[e] = v;
  }
  for (int e = tid; e < ENCO * RDIM; e += 192) W[e] = AtL[e];
  __syncthreads();
  // Gauss-Jordan, thread==row (no cross-thread writes, no pivot array)
  for (int p = 0; p < 128; ++p) {
    if (tid < 128 && tid != p) {
      float f = Gs[tid * 128 + p] / Gs[p * 128 + p];
      for (int c = p; c < 128; ++c) Gs[tid * 128 + c] -= f * Gs[p * 128 + c];
      for (int c = 0; c < 64; ++c) W[tid * 64 + c] -= f * W[p * 64 + c];
    }
    __syncthreads();
  }
  if (tid < 128) {
    float inv = 1.f / Gs[tid * 128 + tid];
    for (int c = 0; c < 64; ++c) W[tid * 64 + c] *= inv;
  }
  __syncthreads();
  // err^2 = ||R0||^2 + sum W*(2*AtR + G0 W)
  float partial = 0.f;
  for (int e = tid; e < ENCO * RDIM; e += 192) {
    int m = e >> 6, c = e & 63;
    float v = 0.f;
    for (int k = 0; k < 128; ++k) v += G[m * 128 + k] * W[k * 64 + c];
    partial += W[e] * (2.f * AtR[e] + v);
  }
  Gs[tid] = partial;
  __syncthreads();
  if (tid == 0) {
    float tot = 0.f;
    for (int t = 0; t < 192; ++t) tot += Gs[t];
    float err2 = scal[1] + tot;
    errs[blockIdx.x] = sqrtf(fmaxf(err2, 0.f)) / sqrtf(scal[0]);
  }
}

// ---------------- nanargmin + outputs ----------------
__global__ void k_final(const float* errs, const float* Wbuf, const float* scal, float* out) {
  __shared__ int sbi;
  const int tid = threadIdx.x;
  if (tid == 0) {
    int bi = 0; float be = 0.f; bool found = false;
    for (int i = 0; i < NLAM; ++i) {
      float e = errs[i];
      if (!(e != e) && (!found || e < be)) { be = e; bi = i; found = true; }
    }
    sbi = bi;
    out[128] = be;
    out[8321] = scal[2];
  }
  __syncthreads();
  const float* W = Wbuf + (size_t)sbi * (ENCO * RDIM);
  for (int e = tid; e < RDIM * ENCO; e += 256) {
    int c = e >> 7, m = e & 127;          // phi_bar[c][m] = W[m][c]
    out[129 + e] = W[m * 64 + c];
  }
}

extern "C" void kernel_launch(void* const* d_in, const int* in_sizes, int n_in,
                              void* d_out, int out_size, void* d_ws, size_t ws_size,
                              hipStream_t stream) {
  (void)in_sizes; (void)n_in; (void)out_size; (void)ws_size;
  const float* X   = (const float*)d_in[0];
  const float* Y   = (const float*)d_in[1];
  const float* lhs = (const float*)d_in[2];
  const float* temperature = (const float*)d_in[3];
  const float* logits = (const float*)d_in[4];
  const float* uni = (const float*)d_in[5];
  const float* U   = (const float*)d_in[6];
  const float* phi = (const float*)d_in[7];
  const float* At  = (const float*)d_in[8];
  const float* lam = (const float*)d_in[9];

  char* ws = (char*)d_ws;
  float* Xt    = (float*)(ws + XT_OFF);
  float* Yt    = (float*)(ws + YT_OFF);
  float* G     = (float*)(ws + G_OFF);
  float* AtL   = (float*)(ws + ATL_OFF);
  float* AtR   = (float*)(ws + ATR_OFF);
  float* errs  = (float*)(ws + ERRS_OFF);
  float* scal  = (float*)(ws + SCAL_OFF);
  float* Wbuf  = (float*)(ws + WBUF_OFF);
  _Float16* bpu   = (_Float16*)(ws + BPU_OFF);
  _Float16* bpphi = (_Float16*)(ws + BPPHI_OFF);
  _Float16* bpat  = (_Float16*)(ws + BPAT_OFF);
  _Float16* bps   = (_Float16*)(ws + BPS_OFF);
  _Float16* tnl   = (_Float16*)(ws + TNL_OFF);
  _Float16* lhsT  = (_Float16*)(ws + LHST_OFF);
  _Float16* r0t   = (_Float16*)(ws + R0T_OFF);
  float* out = (float*)d_out;

  k_zero  <<<64, 256, 0, stream>>>(G, AtL, AtR, scal);
  k_pack  <<<dim3(1024, 4), 256, 0, stream>>>(U, phi, At, lhs, bpu, bpphi, bpat, lhsT);
  k_proj  <<<dim3(32, 2), 256, 0, stream>>>(X, Y, bpu, Xt, Yt, scal);
  k_gumbel<<<128, 256, 0, stream>>>(logits, uni, temperature, bps, out, scal);
  k_r0    <<<32, 256, 0, stream>>>(Xt, Yt, bpat, r0t, scal);
  k_xhatnl<<<64, 128, 0, stream>>>(Xt, bpphi, bps, tnl);
  k_atprod<<<dim3(8, 2), 256, 0, stream>>>(tnl, lhsT, r0t, G, AtL, AtR);
  k_solve <<<16, 192, 0, stream>>>(G, AtL, AtR, lam, Wbuf, scal, errs);
  k_final <<<1, 256, 0, stream>>>(errs, Wbuf, scal, out);
}
